// AssociativeMemory_32323923870430
// MI455X (gfx1250) — matrix operand order repaired and checked by reading, not compiled, not run on hardware
//
#include <hip/hip_runtime.h>
#include <hip/hip_bf16.h>
#include <stdint.h>

// ---------------------------------------------------------------------------
// AssociativeMemory for MI455X (gfx1250, wave32)
//   sim = l2norm(Q) @ l2norm(K)^T * |temp|   (bf16 WMMA, f32 accum)
//   streaming top-32 per row (sim never leaves LDS), softmax,
//   value gather, residual + LayerNorm.
//
// Round-2 change: 4 key tiles (64 keys) per wave iteration -> 4 WMMAs per
// A-fragment load (was 2), halving redundant A traffic and raising the
// WMMA:VMEM issue ratio; prefetch next key tile-group.
// ---------------------------------------------------------------------------

#define DDIM   1024
#define TOPK   32
#define WAVES  4
#define TPB    (WAVES * 32)
#define KTILES 4                  // 16-key WMMA tiles per wave iteration
#define KEYS_PER_ITER (KTILES * 16)
#define NEGINF (-3.0e38f)

typedef __attribute__((ext_vector_type(16))) __bf16 v16bf;
typedef __attribute__((ext_vector_type(8)))  float  v8f;

struct U32x8 { uint4 lo, hi; };

__device__ __forceinline__ v16bf load_frag16(const uint16_t* p0, const uint16_t* p1) {
    U32x8 t;
    t.lo = *(const uint4*)p0;   // 8 bf16
    t.hi = *(const uint4*)p1;   // 8 bf16
    return __builtin_bit_cast(v16bf, t);
}

__device__ __forceinline__ uint32_t f2bf(float f) {
    uint32_t u = __float_as_uint(f);
    return (u + 0x7FFFu + ((u >> 16) & 1u)) >> 16;   // round-to-nearest-even
}

// --------------------------- pass 1/2: L2 normalize -> bf16 -----------------
__global__ __launch_bounds__(256) void l2norm_to_bf16(
    const float* __restrict__ src, uint16_t* __restrict__ dst)
{
    const int row = blockIdx.x;
    const int tid = threadIdx.x;
    const float4* s4 = (const float4*)(src + (size_t)row * DDIM);
    float4 v = s4[tid];                                   // 256 * 4 = 1024
    float ss = v.x * v.x + v.y * v.y + v.z * v.z + v.w * v.w;
    #pragma unroll
    for (int off = 16; off >= 1; off >>= 1) ss += __shfl_xor(ss, off, 32);

    __shared__ float wsum[8];
    const int wid = tid >> 5, lid = tid & 31;
    if (lid == 0) wsum[wid] = ss;
    __syncthreads();
    float tot = 0.f;
    #pragma unroll
    for (int i = 0; i < 8; ++i) tot += wsum[i];

    const float scale = 1.0f / fmaxf(sqrtf(tot), 1e-12f);   // F.normalize eps
    uint32_t p0 = f2bf(v.x * scale) | (f2bf(v.y * scale) << 16);
    uint32_t p1 = f2bf(v.z * scale) | (f2bf(v.w * scale) << 16);
    ((uint2*)(dst + (size_t)row * DDIM))[tid] = make_uint2(p0, p1);
}

// --------------------------- main fused kernel ------------------------------
__global__ __launch_bounds__(TPB) void assoc_main(
    const uint16_t* __restrict__ qn, const uint16_t* __restrict__ kn,
    const float* __restrict__ values, const float* __restrict__ query,
    const float* __restrict__ temperature,
    const float* __restrict__ ln_gamma, const float* __restrict__ ln_beta,
    float* __restrict__ out, float* __restrict__ outAttn, int M)
{
    __shared__ float simLds[WAVES * 16 * KEYS_PER_ITER]; // 16 KB
    __shared__ float listScore[TPB * TOPK];              // 16 KB
    __shared__ int   listIdx[TPB * TOPK];                // 16 KB
    __shared__ float topS[16 * TOPK];                    // 2 KB
    __shared__ int   topI[16 * TOPK];                    // 2 KB
    __shared__ float attnLds[16 * TOPK];                 // 2 KB

    const int tid   = threadIdx.x;
    const int wave  = tid >> 5;
    const int lane  = tid & 31;
    const int lo16  = lane & 15;
    const int hi    = lane >> 4;
    const int qbase = blockIdx.x * 16;
    const float tAbs = fabsf(temperature[0]);

    // init private top-32 list
    float minVal = NEGINF; int minPos = 0;
    #pragma unroll
    for (int p = 0; p < TOPK; ++p) {
        listScore[tid * TOPK + p] = NEGINF;
        listIdx[tid * TOPK + p]   = 0;
    }

    // A fragment (16-bit A 16x32): lane -> M=lo16;
    // VGPR0-3: K = hi*8 + 0..7 ; VGPR4-7: K = 16 + hi*8 + 0..7
    const uint16_t* aRow = qn + (size_t)(qbase + lo16) * DDIM;
    const int aO0 = hi * 8;
    const int aO1 = 16 + hi * 8;
    // B fragment (16-bit B 32x16): lane -> N=lo16;
    // VGPR0-3: K = hi*8 + 0..7 ; VGPR4-7: K = 16 + hi*8 + 0..7
    const int bO = hi * 8;

    const int keysPerWave = M / WAVES;
    const int iters = keysPerWave / KEYS_PER_ITER;
    const int waveKeyBase = wave * keysPerWave;
    float* simW = simLds + wave * (16 * KEYS_PER_ITER);

    for (int it = 0; it < iters; ++it) {
        const int kt = waveKeyBase + it * KEYS_PER_ITER;
        const uint16_t* bRow[KTILES];
        #pragma unroll
        for (int t = 0; t < KTILES; ++t)
            bRow[t] = kn + (size_t)(kt + t * 16 + lo16) * DDIM;

        // speculative prefetch of next tile-group (keys are L2-resident)
        if (it + 1 < iters) {
            #pragma unroll
            for (int t = 0; t < KTILES; ++t)
                __builtin_prefetch(bRow[t] + (size_t)KEYS_PER_ITER * DDIM, 0, 0);
        }

        v8f c[KTILES];
        #pragma unroll
        for (int t = 0; t < KTILES; ++t) c[t] = (v8f){0,0,0,0,0,0,0,0};

        #pragma unroll 2
        for (int kk = 0; kk < DDIM; kk += 32) {
            v16bf a = load_frag16(aRow + kk + aO0, aRow + kk + aO1);
            #pragma unroll
            for (int t = 0; t < KTILES; ++t) {
                v16bf b = load_frag16(bRow[t] + kk + bO, bRow[t] + kk + bO + 16);
                c[t] = __builtin_amdgcn_wmma_f32_16x16x32_bf16(
                           false, a, false, b, (short)0, c[t], false, false);
            }
        }

        // C layout: element v -> M = v + 8*hi, N = lo16 within tile t
        #pragma unroll
        for (int v = 0; v < 8; ++v) {
            const int m = v + 8 * hi;
            #pragma unroll
            for (int t = 0; t < KTILES; ++t)
                simW[m * KEYS_PER_ITER + t * 16 + lo16] = c[t][v];
        }

        // scan: lane owns (row = lo16, half = hi) -> 32 candidates / iter.
        // in-wave LDS write->read: DS ops issue & complete in order per wave.
        for (int j = 0; j < KEYS_PER_ITER / 2; ++j) {
            const int nn = hi * (KEYS_PER_ITER / 2) + j;
            const float val = simW[lo16 * KEYS_PER_ITER + nn] * tAbs;
            if (val > minVal) {
                listScore[tid * TOPK + minPos] = val;
                listIdx[tid * TOPK + minPos]   = kt + nn;
                minVal = 3.0e38f;
                #pragma unroll
                for (int p = 0; p < TOPK; ++p) {
                    const float s = listScore[tid * TOPK + p];
                    if (s < minVal) { minVal = s; minPos = p; }
                }
            }
        }
    }
    __syncthreads();

    // ---- merge: wave w owns rows 4w..4w+3; row r's candidates live in the
    //      8 lane-lists g = r + 16q (q=0..7) -> 256 candidates, 8 per lane
    for (int rr = 0; rr < 4; ++rr) {
        const int r = wave * 4 + rr;
        for (int round = 0; round < TOPK; ++round) {
            float bv = NEGINF; int bp = 0;
            #pragma unroll
            for (int s = 0; s < 8; ++s) {
                const int c2 = lane * 8 + s;
                const int li = (r + 16 * (c2 >> 5)) * TOPK + (c2 & 31);
                const float v = listScore[li];
                if (v > bv) { bv = v; bp = c2; }
            }
            #pragma unroll
            for (int off = 16; off >= 1; off >>= 1) {   // wave32 argmax
                const float ov = __shfl_xor(bv, off, 32);
                const int   op = __shfl_xor(bp, off, 32);
                if (ov > bv || (ov == bv && op < bp)) { bv = ov; bp = op; }
            }
            const int li = (r + 16 * (bp >> 5)) * TOPK + (bp & 31);
            if (lane == (bp >> 3)) listScore[li] = NEGINF;   // consume winner
            if (lane == 0) {
                topS[r * TOPK + round] = bv;
                topI[r * TOPK + round] = listIdx[li];
            }
        }
    }

    // ---- softmax over top-32 (sorted desc, element 0 is the row max)
    for (int rr = 0; rr < 4; ++rr) {
        const int r = wave * 4 + rr;
        const float s  = topS[r * TOPK + lane];
        const float mx = topS[r * TOPK + 0];
        const float e  = expf(s - mx);
        float sum = e;
        #pragma unroll
        for (int off = 16; off >= 1; off >>= 1) sum += __shfl_xor(sum, off, 32);
        const float a = e / sum;
        attnLds[r * TOPK + lane] = a;
        outAttn[(size_t)(qbase + r) * TOPK + lane] = a;
    }

    // ---- value gather + residual + LayerNorm (fp32, per-wave per-row)
    for (int rr = 0; rr < 4; ++rr) {
        const int r = wave * 4 + rr;
        float4 acc[8];
        #pragma unroll
        for (int t = 0; t < 8; ++t) acc[t] = make_float4(0.f, 0.f, 0.f, 0.f);
        for (int k = 0; k < TOPK; ++k) {
            const float w  = attnLds[r * TOPK + k];
            const int   id = topI[r * TOPK + k];
            const float4* vp = (const float4*)(values + (size_t)id * DDIM);
            #pragma unroll
            for (int t = 0; t < 8; ++t) {
                const float4 vv = vp[lane + 32 * t];
                acc[t].x += w * vv.x; acc[t].y += w * vv.y;
                acc[t].z += w * vv.z; acc[t].w += w * vv.w;
            }
        }
        const float4* qp = (const float4*)(query + (size_t)(qbase + r) * DDIM);
        float4 x[8];
        float sum = 0.f, ssq = 0.f;
        #pragma unroll
        for (int t = 0; t < 8; ++t) {
            const float4 qv = qp[lane + 32 * t];
            x[t].x = acc[t].x + qv.x; x[t].y = acc[t].y + qv.y;
            x[t].z = acc[t].z + qv.z; x[t].w = acc[t].w + qv.w;
            sum += x[t].x + x[t].y + x[t].z + x[t].w;
            ssq += x[t].x * x[t].x + x[t].y * x[t].y
                 + x[t].z * x[t].z + x[t].w * x[t].w;
        }
        #pragma unroll
        for (int off = 16; off >= 1; off >>= 1) {
            sum += __shfl_xor(sum, off, 32);
            ssq += __shfl_xor(ssq, off, 32);
        }
        const float mu  = sum * (1.0f / DDIM);
        const float var = fmaxf(ssq * (1.0f / DDIM) - mu * mu, 0.f);
        const float inv = rsqrtf(var + 1e-5f);
        const float4* g4 = (const float4*)ln_gamma;
        const float4* b4 = (const float4*)ln_beta;
        float4* o4 = (float4*)(out + (size_t)(qbase + r) * DDIM);
        #pragma unroll
        for (int t = 0; t < 8; ++t) {
            const int d4 = lane + 32 * t;
            const float4 g = g4[d4], b = b4[d4];
            float4 o;
            o.x = (x[t].x - mu) * inv * g.x + b.x;
            o.y = (x[t].y - mu) * inv * g.y + b.y;
            o.z = (x[t].z - mu) * inv * g.z + b.z;
            o.w = (x[t].w - mu) * inv * g.w + b.w;
            o4[d4] = o;
        }
    }
}

// --------------------------- launch -----------------------------------------
extern "C" void kernel_launch(void* const* d_in, const int* in_sizes, int n_in,
                              void* d_out, int out_size, void* d_ws, size_t ws_size,
                              hipStream_t stream) {
    const float* query = (const float*)d_in[0];
    const float* keys  = (const float*)d_in[1];
    const float* vals  = (const float*)d_in[2];
    const float* temp  = (const float*)d_in[3];
    const float* gamma = (const float*)d_in[4];
    const float* beta  = (const float*)d_in[5];

    const int N = in_sizes[0] / DDIM;   // 8192
    const int M = in_sizes[1] / DDIM;   // 32768

    // workspace: bf16 qn [N,D] (16 MB) then bf16 kn [M,D] (64 MB)
    uint16_t* qn = (uint16_t*)d_ws;
    uint16_t* kn = qn + (size_t)N * DDIM;

    float* out     = (float*)d_out;
    float* outAttn = out + (size_t)N * DDIM;

    l2norm_to_bf16<<<N, 256, 0, stream>>>(query, qn);
    l2norm_to_bf16<<<M, 256, 0, stream>>>(keys, kn);
    assoc_main<<<N / 16, TPB, 0, stream>>>(qn, kn, vals, query, temp,
                                           gamma, beta, out, outAttn, M);
}
